// ACE_88991722373533
// MI455X (gfx1250) — matrix operand order repaired and checked by reading, not compiled, not run on hardware
//
#include <hip/hip_runtime.h>

// ---------------------------------------------------------------------------
// CDNA5 (gfx1250, wave32) fused autoencoder + cluster-softmin pipeline.
// Pre-pass converts x / weights to f16 (weights transposed to [N,K]) so every
// GEMM operand is K-major f16: staging is pure b128 copies (async-to-LDS when
// the toolchain exposes the gfx1250 builtin), WMMA f32 accumulation, fused
// bias/ReLU/softmin epilogues. Double-buffered LDS, 80B pitch, wave32 layouts.
// ---------------------------------------------------------------------------

typedef _Float16 v4h  __attribute__((ext_vector_type(4)));
typedef _Float16 v8h  __attribute__((ext_vector_type(8)));
typedef _Float16 v16h __attribute__((ext_vector_type(16)));
typedef float    v4f  __attribute__((ext_vector_type(4)));
typedef float    v8f  __attribute__((ext_vector_type(8)));
typedef int      v4i  __attribute__((ext_vector_type(4)));

#define BM 128
#define BN 128
#define BK 32
#define LDP 40   // LDS row pitch in halfs: 80B = 16B-aligned, 20-word bank stride

// -------- gfx1250 async load-to-LDS (guarded; full fallback provided) -------
#if defined(__has_builtin)
#  if __has_builtin(__builtin_amdgcn_global_load_async_to_lds_b128) && \
      __has_builtin(__builtin_amdgcn_s_wait_asynccnt)
#    define ACE_ASYNC 1
#  endif
#endif
#ifndef ACE_ASYNC
#  define ACE_ASYNC 0
#endif

#if ACE_ASYNC
typedef __attribute__((address_space(1))) v4i glb_v4i;
typedef __attribute__((address_space(3))) v4i lds_v4i;
__device__ __forceinline__ void ace_async_b128(const _Float16* g, _Float16* l) {
    __builtin_amdgcn_global_load_async_to_lds_b128((glb_v4i*)g, (lds_v4i*)l, 0, 0);
}
#endif

union frag16 {
    v16h v;
    struct { v8h lo; v8h hi; } p;
};

// ---------------------------------------------------------------------------
// Pre-pass: f32 -> f16 convert (vectorized), and f32 [K,N] -> f16 [N,K].
// ---------------------------------------------------------------------------
__global__ __launch_bounds__(256)
void ACE_cvt_f16(const float* __restrict__ s, _Float16* __restrict__ d, long n)
{
    long i = ((long)blockIdx.x * blockDim.x + threadIdx.x) * 4;
    if (i + 3 < n) {
        v4f f = *(const v4f*)(s + i);
        v4h h = { (_Float16)f[0], (_Float16)f[1], (_Float16)f[2], (_Float16)f[3] };
        *(v4h*)(d + i) = h;
    }
}

__global__ __launch_bounds__(256)
void ACE_cvt_transpose(const float* __restrict__ S /*[K,N]*/,
                       _Float16* __restrict__ D /*[N,K]*/, int K, int N)
{
    __shared__ _Float16 t[32][33];
    const int tx = threadIdx.x & 31, ty = threadIdx.x >> 5;   // 32 x 8
    const int k0 = blockIdx.y * 32, n0 = blockIdx.x * 32;
#pragma unroll
    for (int i = 0; i < 32; i += 8)
        t[ty + i][tx] = (_Float16)S[(long)(k0 + ty + i) * N + n0 + tx];
    __syncthreads();
#pragma unroll
    for (int i = 0; i < 32; i += 8)
        D[(long)(n0 + ty + i) * K + k0 + tx] = t[tx][ty + i];
}

// ---------------------------------------------------------------------------
// C = act(A[M,K]_f16 @ Bt[N,K]_f16^T + bias[N]); optional f32 and f16 outputs.
// 256 threads = 8 waves (4 row x 2 col); block tile 128x128; wave 32x64
// (2x4 16x16x32 WMMA); K-step 32; double-buffered staging, 1 barrier/iter.
// ---------------------------------------------------------------------------
__global__ __launch_bounds__(256)
void ACE_gemm_f16(const _Float16* __restrict__ A, const _Float16* __restrict__ Bt,
                  const float* __restrict__ bias,
                  float* __restrict__ Cf, _Float16* __restrict__ Ch,
                  int M, int N, int K, int relu)
{
    __shared__ _Float16 sA[2][BM * LDP];
    __shared__ _Float16 sB[2][BN * LDP];   // [n][k]

    const int tid     = threadIdx.x;
    const int lane    = tid & 31;
    const int wid     = tid >> 5;
    const int wr      = wid & 3;           // wave row 0..3  (32 rows each)
    const int wc      = wid >> 2;          // wave col 0..1  (64 cols each)
    const int l16     = lane & 15;
    const int halfSel = lane >> 4;
    const int mBlk    = blockIdx.y * BM;
    const int nBlk    = blockIdx.x * BN;

    // staging: tile row = 32 halfs = 64B = 4 b128 lane-slots; 2 passes of 64 rows
    const int sRow = tid >> 2;             // 0..63 (+64 on pass 1)
    const int sCol = (tid & 3) * 8;        // half offset within row

    const _Float16* aPtr = A  + (long)(mBlk + sRow) * K + sCol;
    const _Float16* bPtr = Bt + (long)(nBlk + sRow) * K + sCol;
    const long rowStep = (long)64 * K;

#if ACE_ASYNC
    auto stageAsync = [&](int buf) {
#pragma unroll
        for (int p = 0; p < 2; ++p) {
            ace_async_b128(aPtr + p * rowStep, &sA[buf][(sRow + 64 * p) * LDP + sCol]);
            ace_async_b128(bPtr + p * rowStep, &sB[buf][(sRow + 64 * p) * LDP + sCol]);
        }
        aPtr += BK;
        bPtr += BK;
    };
#else
    v8h aReg[2], bReg[2];
    auto loadTile = [&]() {
#pragma unroll
        for (int p = 0; p < 2; ++p) {
            aReg[p] = *(const v8h*)(aPtr + p * rowStep);
            bReg[p] = *(const v8h*)(bPtr + p * rowStep);
        }
        aPtr += BK;
        bPtr += BK;
    };
    auto stageTile = [&](int buf) {
#pragma unroll
        for (int p = 0; p < 2; ++p) {
            *(v8h*)&sA[buf][(sRow + 64 * p) * LDP + sCol] = aReg[p];
            *(v8h*)&sB[buf][(sRow + 64 * p) * LDP + sCol] = bReg[p];
        }
    };
#endif

    v8f acc[2][4];
#pragma unroll
    for (int r = 0; r < 2; ++r)
#pragma unroll
        for (int c = 0; c < 4; ++c)
            acc[r][c] = (v8f)0.0f;

#if ACE_ASYNC
    stageAsync(0);
    __builtin_amdgcn_s_wait_asynccnt(0);
    __syncthreads();
#else
    loadTile();
    stageTile(0);
    __syncthreads();
#endif

    int buf = 0;
    for (int k0 = 0; k0 < K; k0 += BK) {
        const bool hasNext = (k0 + BK) < K;
#if ACE_ASYNC
        if (hasNext) stageAsync(buf ^ 1);    // async engine fills other buffer
#else
        if (hasNext) loadTile();             // loads in flight over WMMAs
#endif
        if (k0 + 2 * BK < K) {               // prefetch 2 tiles ahead into L2
            __builtin_prefetch(aPtr, 0, 0);
            __builtin_prefetch(bPtr, 0, 0);
        }

        frag16 aF[2], bF[4];
        const _Float16* __restrict__ pA = sA[buf];
        const _Float16* __restrict__ pB = sB[buf];
#pragma unroll
        for (int r = 0; r < 2; ++r) {
            // A 16x32: lanes 0-15 K 0-15; lanes 16-31 K 16-31
            const _Float16* base = &pA[(wr * 32 + r * 16 + l16) * LDP + halfSel * 16];
            aF[r].p.lo = *(const v8h*)(base);
            aF[r].p.hi = *(const v8h*)(base + 8);
        }
#pragma unroll
        for (int c = 0; c < 4; ++c) {
            // B 32x16: lanes 0-15 K 0-15; lanes 16-31 K 16-31
            const _Float16* base = &pB[(wc * 64 + c * 16 + l16) * LDP + halfSel * 16];
            bF[c].p.lo = *(const v8h*)(base);
            bF[c].p.hi = *(const v8h*)(base + 8);
        }

#pragma unroll
        for (int r = 0; r < 2; ++r)
#pragma unroll
            for (int c = 0; c < 4; ++c)
                acc[r][c] = __builtin_amdgcn_wmma_f32_16x16x32_f16(
                    false, aF[r].v, false, bF[c].v,
                    (short)0, acc[r][c], false, false);

#if ACE_ASYNC
        if (hasNext) __builtin_amdgcn_s_wait_asynccnt(0);
#else
        if (hasNext) stageTile(buf ^ 1);
#endif
        __syncthreads();
        buf ^= 1;
    }

    // ---- epilogue: bias + optional ReLU; f32 and/or f16 outputs ----
    float bv[4];
#pragma unroll
    for (int c = 0; c < 4; ++c)
        bv[c] = bias[nBlk + wc * 64 + c * 16 + l16];

#pragma unroll
    for (int r = 0; r < 2; ++r) {
#pragma unroll
        for (int c = 0; c < 4; ++c) {
            const int col = nBlk + wc * 64 + c * 16 + l16;
#pragma unroll
            for (int i = 0; i < 8; ++i) {
                const int row = mBlk + wr * 32 + r * 16 + i + halfSel * 8;
                float v = acc[r][c][i] + bv[c];
                if (relu) v = fmaxf(v, 0.0f);
                if (Cf) Cf[(long)row * N + col] = v;
                if (Ch) Ch[(long)row * N + col] = (_Float16)v;
            }
        }
    }
}

// ---------------------------------------------------------------------------
// conf[b,c] = -logsumexp_k(-10 * (<z_b, clus_w[c,k,:]> + clus_b[c,k]))
// z and clus_w pre-converted to f16 (clus_w[c] is already [n][k] K-major).
// 8 waves x (16 rows x 64 heads); softmin fused in-register.
// ---------------------------------------------------------------------------
__global__ __launch_bounds__(256)
void ACE_cluster_conf(const _Float16* __restrict__ zh, const _Float16* __restrict__ cwh,
                      const float* __restrict__ cb, float* __restrict__ conf, int B)
{
    __shared__ _Float16 sZ[2][128 * LDP];
    __shared__ _Float16 sW[2][64 * LDP];

    const int tid     = threadIdx.x;
    const int lane    = tid & 31;
    const int wid     = tid >> 5;
    const int l16     = lane & 15;
    const int halfSel = lane >> 4;
    const int c       = blockIdx.y;
    const int mBlk    = blockIdx.x * 128;

    const int sRow = tid >> 2;           // 0..63
    const int sCol = (tid & 3) * 8;

    const _Float16* zPtr = zh  + (long)(mBlk + sRow) * 256 + sCol;
    const _Float16* wPtr = cwh + ((long)c * 64 + sRow) * 256 + sCol;  // sRow<64 rows

#if ACE_ASYNC
    auto stageAsync = [&](int buf) {
#pragma unroll
        for (int p = 0; p < 2; ++p)
            ace_async_b128(zPtr + (long)p * 64 * 256,
                           &sZ[buf][(sRow + 64 * p) * LDP + sCol]);
        ace_async_b128(wPtr, &sW[buf][sRow * LDP + sCol]);
        zPtr += BK;
        wPtr += BK;
    };
#else
    v8h zReg[2], wReg;
    auto loadTile = [&]() {
#pragma unroll
        for (int p = 0; p < 2; ++p)
            zReg[p] = *(const v8h*)(zPtr + (long)p * 64 * 256);
        wReg = *(const v8h*)(wPtr);
        zPtr += BK;
        wPtr += BK;
    };
    auto stageTile = [&](int buf) {
#pragma unroll
        for (int p = 0; p < 2; ++p)
            *(v8h*)&sZ[buf][(sRow + 64 * p) * LDP + sCol] = zReg[p];
        *(v8h*)&sW[buf][sRow * LDP + sCol] = wReg;
    };
#endif

    v8f acc[4];
#pragma unroll
    for (int j = 0; j < 4; ++j) acc[j] = (v8f)0.0f;

#if ACE_ASYNC
    stageAsync(0);
    __builtin_amdgcn_s_wait_asynccnt(0);
    __syncthreads();
#else
    loadTile();
    stageTile(0);
    __syncthreads();
#endif

    int buf = 0;
    for (int k0 = 0; k0 < 256; k0 += BK) {
        const bool hasNext = (k0 + BK) < 256;
#if ACE_ASYNC
        if (hasNext) stageAsync(buf ^ 1);
#else
        if (hasNext) loadTile();
#endif

        frag16 aF, bF[4];
        const _Float16* __restrict__ pZ = sZ[buf];
        const _Float16* __restrict__ pW = sW[buf];
        {
            const _Float16* base = &pZ[(wid * 16 + l16) * LDP + halfSel * 16];
            aF.p.lo = *(const v8h*)(base);
            aF.p.hi = *(const v8h*)(base + 8);
        }
#pragma unroll
        for (int j = 0; j < 4; ++j) {
            const _Float16* base = &pW[(j * 16 + l16) * LDP + halfSel * 16];
            bF[j].p.lo = *(const v8h*)(base);
            bF[j].p.hi = *(const v8h*)(base + 8);
        }
#pragma unroll
        for (int j = 0; j < 4; ++j)
            acc[j] = __builtin_amdgcn_wmma_f32_16x16x32_f16(
                false, aF.v, false, bF[j].v, (short)0, acc[j], false, false);

#if ACE_ASYNC
        if (hasNext) __builtin_amdgcn_s_wait_asynccnt(0);
#else
        if (hasNext) stageTile(buf ^ 1);
#endif
        __syncthreads();
        buf ^= 1;
    }

    float bv[4];
#pragma unroll
    for (int j = 0; j < 4; ++j)
        bv[j] = cb[c * 64 + j * 16 + l16];

#pragma unroll
    for (int i = 0; i < 8; ++i) {
        float t0 = -10.0f * (acc[0][i] + bv[0]);
        float t1 = -10.0f * (acc[1][i] + bv[1]);
        float t2 = -10.0f * (acc[2][i] + bv[2]);
        float t3 = -10.0f * (acc[3][i] + bv[3]);
        float m = fmaxf(fmaxf(t0, t1), fmaxf(t2, t3));
#pragma unroll
        for (int off = 1; off < 16; off <<= 1)
            m = fmaxf(m, __shfl_xor(m, off, 32));
        float s = __expf(t0 - m) + __expf(t1 - m) + __expf(t2 - m) + __expf(t3 - m);
#pragma unroll
        for (int off = 1; off < 16; off <<= 1)
            s += __shfl_xor(s, off, 32);
        if (l16 == 0) {
            const int row = mBlk + wid * 16 + i + halfSel * 8;
            conf[(long)row * 64 + c] = -(m + __logf(s));
        }
    }
}

// ---------------------------------------------------------------------------
extern "C" void kernel_launch(void* const* d_in, const int* in_sizes, int n_in,
                              void* d_out, int out_size, void* d_ws, size_t ws_size,
                              hipStream_t stream)
{
    (void)in_sizes; (void)n_in; (void)out_size; (void)ws_size;

    const float* x   = (const float*)d_in[0];
    const float* ew0 = (const float*)d_in[1];
    const float* eb0 = (const float*)d_in[2];
    const float* ew1 = (const float*)d_in[3];
    const float* eb1 = (const float*)d_in[4];
    const float* ew2 = (const float*)d_in[5];
    const float* eb2 = (const float*)d_in[6];
    const float* dw0 = (const float*)d_in[7];
    const float* db0 = (const float*)d_in[8];
    const float* dw1 = (const float*)d_in[9];
    const float* db1 = (const float*)d_in[10];
    const float* dw2 = (const float*)d_in[11];
    const float* db2 = (const float*)d_in[12];
    const float* cw  = (const float*)d_in[13];
    const float* cb  = (const float*)d_in[14];

    const int B = 8192;
    float* out   = (float*)d_out;
    float* zbuf  = out;                               // [B,256]  f32
    float* recon = out + (size_t)B * 256;             // [B,1024] f32
    float* conf  = out + (size_t)B * (256 + 1024);    // [B,64]   f32

    // ---- f16 workspace layout ----
    _Float16* w = (_Float16*)d_ws;
    size_t off = 0;
    auto alloc = [&](size_t n) { _Float16* p = w + off; off += n; return p; };
    _Float16* xh   = alloc((size_t)B * 1024);
    _Float16* h1h  = alloc((size_t)B * 2048);   // reused for d1
    _Float16* h2h  = alloc((size_t)B * 1024);   // reused for d0
    _Float16* zh   = alloc((size_t)B * 256);
    _Float16* ew0t = alloc((size_t)2048 * 1024);
    _Float16* ew1t = alloc((size_t)1024 * 2048);
    _Float16* ew2t = alloc((size_t)256 * 1024);
    _Float16* dw0t = alloc((size_t)1024 * 256);
    _Float16* dw1t = alloc((size_t)2048 * 1024);
    _Float16* dw2t = alloc((size_t)1024 * 2048);
    _Float16* cwh  = alloc((size_t)64 * 64 * 256);

    dim3 blk(256);
    auto cgrd = [](long n) { return dim3((unsigned)((n / 4 + 255) / 256)); };
    auto tgrd = [](int K, int N) { return dim3(N / 32, K / 32); };
    auto grd  = [](int M, int N) { return dim3(N / BN, M / BM); };

    // ---- pre-pass: f16 conversion (+ weight transposes to [N,K]) ----
    ACE_cvt_f16<<<cgrd((long)B * 1024), blk, 0, stream>>>(x, xh, (long)B * 1024);
    ACE_cvt_f16<<<cgrd((long)64 * 64 * 256), blk, 0, stream>>>(cw, cwh, (long)64 * 64 * 256);
    ACE_cvt_transpose<<<tgrd(1024, 2048), blk, 0, stream>>>(ew0, ew0t, 1024, 2048);
    ACE_cvt_transpose<<<tgrd(2048, 1024), blk, 0, stream>>>(ew1, ew1t, 2048, 1024);
    ACE_cvt_transpose<<<tgrd(1024,  256), blk, 0, stream>>>(ew2, ew2t, 1024,  256);
    ACE_cvt_transpose<<<tgrd( 256, 1024), blk, 0, stream>>>(dw0, dw0t,  256, 1024);
    ACE_cvt_transpose<<<tgrd(1024, 2048), blk, 0, stream>>>(dw1, dw1t, 1024, 2048);
    ACE_cvt_transpose<<<tgrd(2048, 1024), blk, 0, stream>>>(dw2, dw2t, 2048, 1024);

    // ---- encoder ----
    ACE_gemm_f16<<<grd(B, 2048), blk, 0, stream>>>(xh,  ew0t, eb0, nullptr, h1h, B, 2048, 1024, 1);
    ACE_gemm_f16<<<grd(B, 1024), blk, 0, stream>>>(h1h, ew1t, eb1, nullptr, h2h, B, 1024, 2048, 1);
    ACE_gemm_f16<<<grd(B,  256), blk, 0, stream>>>(h2h, ew2t, eb2, zbuf,    zh,  B,  256, 1024, 0);
    // ---- decoder (h2h reused as d0, h1h reused as d1) ----
    ACE_gemm_f16<<<grd(B, 1024), blk, 0, stream>>>(zh,  dw0t, db0, nullptr, h2h, B, 1024,  256, 1);
    ACE_gemm_f16<<<grd(B, 2048), blk, 0, stream>>>(h2h, dw1t, db1, nullptr, h1h, B, 2048, 1024, 1);
    ACE_gemm_f16<<<grd(B, 1024), blk, 0, stream>>>(h1h, dw2t, db2, recon, nullptr, B, 1024, 2048, 0);
    // ---- fused cluster heads + softmin pool ----
    ACE_cluster_conf<<<dim3(B / 128, 64), blk, 0, stream>>>(zh, cwh, cb, conf, B);
}
